// DemoRecModel_41145786696440
// MI455X (gfx1250) — matrix operand order repaired and checked by reading, not compiled, not run on hardware
//
#include <hip/hip_runtime.h>
#include <hip/hip_bf16.h>

// ---------------- problem constants ----------------
constexpr int kB  = 256;
constexpr int kL  = 200;
constexpr int kH  = 256;
constexpr int kNH = 4;
constexpr int kNL = 2;
constexpr int kFF = 1024;
constexpr int kDH = kH / kNH;          // 64
constexpr int kM  = kB * kL;           // 51200 rows, divisible by 32
constexpr int kLP = 224;               // L padded to multiple of 32 (7 WMMA k-steps)

// ---------------- vector types ----------------
typedef __attribute__((ext_vector_type(16))) __bf16 bv16;
typedef __attribute__((ext_vector_type(8)))  float  v8f;

static __device__ __forceinline__ v8f wmma_bf16(bv16 a, bv16 b, v8f c) {
    // v_wmma_f32_16x16x32_bf16 : D = A(16x32) * B(32x16) + C
    return __builtin_amdgcn_wmma_f32_16x16x32_bf16(false, a, false, b,
                                                   (short)0, c, false, false);
}

// Load a 32-byte fragment from two 16-byte-aligned halves.
static __device__ __forceinline__ bv16 load_frag(const __bf16* p0, const __bf16* p1) {
    bv16 r;
    *reinterpret_cast<uint4*>(&r)       = *reinterpret_cast<const uint4*>(p0);
    *(reinterpret_cast<uint4*>(&r) + 1) = *reinterpret_cast<const uint4*>(p1);
    return r;
}

// ---------------- weight pack: W[K,N] fp32 -> Wt[N,K] bf16 ----------------
__global__ void transpose_w_kernel(const float* __restrict__ W,
                                   __bf16* __restrict__ Wt, int K, int N) {
    int i = blockIdx.x * 256 + threadIdx.x;   // over N*K
    if (i >= N * K) return;
    int n = i / K, k = i % K;
    Wt[i] = (__bf16)W[(size_t)k * N + n];
}

// ---------------- embed + LayerNorm ----------------
__global__ __launch_bounds__(256)
void embed_ln_kernel(const int* __restrict__ ids,
                     const float* __restrict__ item_emb,
                     const float* __restrict__ pos_emb,
                     const float* __restrict__ w, const float* __restrict__ bln,
                     float* __restrict__ X, __bf16* __restrict__ Xh) {
    int row = blockIdx.x;            // b*L + l
    int l   = row % kL;
    int t   = threadIdx.x;
    int id  = ids[row];
    float v = item_emb[(size_t)id * kH + t] + pos_emb[(size_t)l * kH + t];

    __shared__ float red[256];
    red[t] = v; __syncthreads();
    for (int s = 128; s > 0; s >>= 1) { if (t < s) red[t] += red[t + s]; __syncthreads(); }
    float mu = red[0] / kH; __syncthreads();
    float d = v - mu;
    red[t] = d * d; __syncthreads();
    for (int s = 128; s > 0; s >>= 1) { if (t < s) red[t] += red[t + s]; __syncthreads(); }
    float var = red[0] / kH;
    float y = w[t] * d * rsqrtf(var + 1e-12f) + bln[t];
    size_t idx = (size_t)row * kH + t;
    X[idx]  = y;
    Xh[idx] = (__bf16)y;
}

// ---------------- residual add + LayerNorm ----------------
__global__ __launch_bounds__(256)
void add_ln_kernel(const float* __restrict__ a, const float* __restrict__ res,
                   const float* __restrict__ w, const float* __restrict__ bln,
                   float* __restrict__ X, __bf16* __restrict__ Xh) {
    int row = blockIdx.x;
    int t   = threadIdx.x;
    size_t idx = (size_t)row * kH + t;
    float v = a[idx] + res[idx];

    __shared__ float red[256];
    red[t] = v; __syncthreads();
    for (int s = 128; s > 0; s >>= 1) { if (t < s) red[t] += red[t + s]; __syncthreads(); }
    float mu = red[0] / kH; __syncthreads();
    float d = v - mu;
    red[t] = d * d; __syncthreads();
    for (int s = 128; s > 0; s >>= 1) { if (t < s) red[t] += red[t + s]; __syncthreads(); }
    float var = red[0] / kH;
    float y = w[t] * d * rsqrtf(var + 1e-12f) + bln[t];
    X[idx]  = y;
    Xh[idx] = (__bf16)y;
}

// ---------------- generic WMMA GEMM: Y = act(A[M,K] * W + bias) ----------------
// A: bf16 [M,K] row-major.  Wt: bf16 [N,K] (= W^T, so B[k,n] contiguous in k).
// Register-blocked 2x2: each wave owns a 32x32 output tile (4 WMMA accumulators),
// reusing every A/B fragment twice -> 2 b128-loads per WMMA instead of 4.
// grid = (M/32, N/(32*GEMM_WAVES)), block = 32*GEMM_WAVES.
#define GEMM_WAVES 4
__global__ __launch_bounds__(32 * GEMM_WAVES)
void gemm_bias_kernel(const __bf16* __restrict__ A, const __bf16* __restrict__ Wt,
                      const float* __restrict__ bias,
                      float* __restrict__ outF, __bf16* __restrict__ outH,
                      int M, int N, int K, int act) {
    const int lane = threadIdx.x & 31;
    const int wave = threadIdx.x >> 5;
    const int tm = blockIdx.x;                          // 32-row tile
    const int tn = blockIdx.y * GEMM_WAVES + wave;      // 32-col tile
    const int g  = lane >> 4;
    const int ln = lane & 15;

    const __bf16* a0Base = A  + (size_t)(tm * 32 + ln)      * K + g * 8;
    const __bf16* a1Base = A  + (size_t)(tm * 32 + 16 + ln) * K + g * 8;
    const __bf16* b0Base = Wt + (size_t)(tn * 32 + ln)      * K + g * 8;
    const __bf16* b1Base = Wt + (size_t)(tn * 32 + 16 + ln) * K + g * 8;

    v8f acc00 = {}, acc01 = {}, acc10 = {}, acc11 = {};
    for (int k = 0; k < K; k += 32) {
        bv16 a0 = load_frag(a0Base + k, a0Base + k + 16);
        bv16 a1 = load_frag(a1Base + k, a1Base + k + 16);
        bv16 b0 = load_frag(b0Base + k, b0Base + k + 16);
        bv16 b1 = load_frag(b1Base + k, b1Base + k + 16);
        acc00 = wmma_bf16(a0, b0, acc00);
        acc01 = wmma_bf16(a0, b1, acc01);
        acc10 = wmma_bf16(a1, b0, acc10);
        acc11 = wmma_bf16(a1, b1, acc11);
    }

    const v8f* accs[4] = { &acc00, &acc01, &acc10, &acc11 };
#pragma unroll
    for (int mi = 0; mi < 2; ++mi) {
#pragma unroll
        for (int ni = 0; ni < 2; ++ni) {
            const v8f& acc = *accs[mi * 2 + ni];
            int colN = tn * 32 + ni * 16 + ln;
            float bvv = bias ? bias[colN] : 0.0f;
#pragma unroll
            for (int r = 0; r < 8; ++r) {
                int row = tm * 32 + mi * 16 + r + 8 * g;
                float y = acc[r] + bvv;
                if (act == 1) y = 0.5f * y * (1.0f + erff(y * 0.70710678118654752f));
                size_t idx = (size_t)row * N + colN;
                if (outF) outF[idx] = y;
                if (outH) outH[idx] = (__bf16)y;
            }
        }
    }
}

// -------- V repack: [B*L,H] bf16 -> Vt[b,h,DH,LP] bf16 (zero-padded keys) --------
__global__ void pack_vt_kernel(const __bf16* __restrict__ V, __bf16* __restrict__ Vt) {
    int i = blockIdx.x * 256 + threadIdx.x;   // over B*NH*DH*LP
    if (i >= kB * kNH * kDH * kLP) return;
    int key = i % kLP;
    int d   = (i / kLP) % kDH;
    int h   = (i / (kLP * kDH)) % kNH;
    int b   = i / (kLP * kDH * kNH);
    Vt[i] = (key < kL) ? V[((size_t)(b * kL + key)) * kH + h * kDH + d] : (__bf16)0.0f;
}

// -------- fused scores (Q K^T) + mask + softmax -> probs bf16 --------
// grid = (LP/16, NH, B), block = 32 (one wave). Wave computes a 16 x 224 score strip
// with 14 WMMA accumulators (contraction over DH=64 -> 2 k-steps), then softmax via LDS.
// Q fragments are loop-invariant over key-tiles (hoisted by the compiler).
__global__ __launch_bounds__(32)
void attn_scores_softmax_kernel(const __bf16* __restrict__ Q, const __bf16* __restrict__ Kb,
                                const int* __restrict__ ids, __bf16* __restrict__ probs) {
    const int qt = blockIdx.x, h = blockIdx.y, b = blockIdx.z;
    const int lane = threadIdx.x;
    const int g = lane >> 4, ln = lane & 15;

    __shared__ float srow[16 * kLP];

    int qRow = qt * 16 + ln;
    int qC   = (qRow < kL) ? qRow : (kL - 1);      // clamp pad rows (discarded later)
    const __bf16* qBase = Q + ((size_t)(b * kL + qC)) * kH + h * kDH + g * 8;

    v8f acc[kLP / 16];
#pragma unroll
    for (int kt = 0; kt < kLP / 16; ++kt) {
        int key = kt * 16 + ln;
        int kC  = (key < kL) ? key : (kL - 1);
        const __bf16* kBase = Kb + ((size_t)(b * kL + kC)) * kH + h * kDH + g * 8;
        v8f c = {};
#pragma unroll
        for (int k = 0; k < kDH; k += 32) {
            bv16 a  = load_frag(qBase + k, qBase + k + 16);
            bv16 bb = load_frag(kBase + k, kBase + k + 16);
            c = wmma_bf16(a, bb, c);
        }
        acc[kt] = c;
    }
#pragma unroll
    for (int kt = 0; kt < kLP / 16; ++kt)
#pragma unroll
        for (int r = 0; r < 8; ++r)
            srow[(r + 8 * g) * kLP + kt * 16 + ln] = acc[kt][r];
    __syncthreads();

    if (lane < 16) {
        int q = qt * 16 + lane;
        float* row = &srow[lane * kLP];
        size_t outBase = (((size_t)(b * kNH + h)) * kLP + q) * kLP;
        if (q < kL) {
            const float scale = 0.125f;  // 1/sqrt(64)
            float mx = -1e30f;
            for (int key = 0; key < kL; ++key) {
                float add = ((key <= q) && (ids[b * kL + key] > 0)) ? 0.0f : -10000.0f;
                float s = row[key] * scale + add;
                row[key] = s;
                mx = fmaxf(mx, s);
            }
            float sum = 0.0f;
            for (int key = 0; key < kL; ++key) {
                float e = expf(row[key] - mx);
                row[key] = e;
                sum += e;
            }
            float inv = 1.0f / sum;
            for (int key = 0; key < kLP; ++key)
                probs[outBase + key] = (key < kL) ? (__bf16)(row[key] * inv) : (__bf16)0.0f;
        } else {
            for (int key = 0; key < kLP; ++key) probs[outBase + key] = (__bf16)0.0f;
        }
    }
}

// -------- ctx = probs @ V  (per b,h; one wave computes 16 x 64, A reused 4x) --------
__global__ __launch_bounds__(32)
void attn_ctx_kernel(const __bf16* __restrict__ probs, const __bf16* __restrict__ Vt,
                     __bf16* __restrict__ ctx) {
    const int qt = blockIdx.x, h = blockIdx.y, b = blockIdx.z;
    const int lane = threadIdx.x;
    const int g = lane >> 4, ln = lane & 15;

    const __bf16* aBase = probs + (((size_t)(b * kNH + h)) * kLP + qt * 16 + ln) * kLP + g * 8;
    const __bf16* vBase = Vt + ((size_t)(b * kNH + h)) * kDH * kLP + g * 8;

    v8f acc[kDH / 16];
#pragma unroll
    for (int nt = 0; nt < kDH / 16; ++nt) { v8f z = {}; acc[nt] = z; }

    for (int k = 0; k < kLP; k += 32) {
        bv16 a = load_frag(aBase + k, aBase + k + 16);
#pragma unroll
        for (int nt = 0; nt < kDH / 16; ++nt) {
            const __bf16* vp = vBase + (size_t)(nt * 16 + ln) * kLP + k;
            bv16 bb = load_frag(vp, vp + 16);
            acc[nt] = wmma_bf16(a, bb, acc[nt]);
        }
    }
#pragma unroll
    for (int nt = 0; nt < kDH / 16; ++nt) {
        int col = h * kDH + nt * 16 + ln;
#pragma unroll
        for (int r = 0; r < 8; ++r) {
            int q = qt * 16 + r + 8 * g;
            if (q < kL) ctx[((size_t)(b * kL + q)) * kH + col] = (__bf16)acc[nt][r];
        }
    }
}

// ---------------- orchestration ----------------
extern "C" void kernel_launch(void* const* d_in, const int* in_sizes, int n_in,
                              void* d_out, int out_size, void* d_ws, size_t ws_size,
                              hipStream_t stream) {
    (void)in_sizes; (void)n_in; (void)out_size; (void)ws_size;
    const float* item_emb  = (const float*)d_in[0];
    const float* pos_emb   = (const float*)d_in[1];
    const float* emb_ln_w  = (const float*)d_in[2];
    const float* emb_ln_b  = (const float*)d_in[3];
    const float* Wq = (const float*)d_in[4];  const float* bq = (const float*)d_in[5];
    const float* Wk = (const float*)d_in[6];  const float* bk = (const float*)d_in[7];
    const float* Wv = (const float*)d_in[8];  const float* bv = (const float*)d_in[9];
    const float* Wo = (const float*)d_in[10]; const float* bo = (const float*)d_in[11];
    const float* attn_ln_w = (const float*)d_in[12];
    const float* attn_ln_b = (const float*)d_in[13];
    const float* W1 = (const float*)d_in[14]; const float* b1 = (const float*)d_in[15];
    const float* W2 = (const float*)d_in[16]; const float* b2 = (const float*)d_in[17];
    const float* ffn_ln_w  = (const float*)d_in[18];
    const float* ffn_ln_b  = (const float*)d_in[19];
    const int*   input_ids = (const int*)d_in[20];
    float* out = (float*)d_out;

    // ---- workspace carve (bytes); FFN hidden aliases dead attention scratch ----
    char* ws = (char*)d_ws;
    const size_t szX   = (size_t)kM * kH * 4;                            // 52,428,800
    const size_t szXh  = (size_t)kM * kH * 2;                            // 26,214,400
    const size_t szWt  = (size_t)kNL * (4 * kH * kH + 2 * kH * kFF) * 2; //  3,145,728
    const size_t szVt  = (size_t)kB * kNH * kDH * kLP * 2;               // 29,360,128
    const size_t szPr  = (size_t)kB * kNH * kLP * kLP * 2;               // 102,760,448
    float*  X     = (float*)(ws);
    __bf16* Xh    = (__bf16*)(ws + szX);
    __bf16* WtAll = (__bf16*)(ws + szX + szXh);
    char*   S     = ws + szX + szXh + szWt;
    __bf16* Qh    = (__bf16*)(S);
    __bf16* Kh    = (__bf16*)(S + szXh);
    __bf16* Vh    = (__bf16*)(S + 2 * szXh);
    __bf16* Vt    = (__bf16*)(S + 3 * szXh);
    __bf16* probs = (__bf16*)(S + 3 * szXh + szVt);
    __bf16* ctx   = (__bf16*)(S + 3 * szXh + szVt + szPr);
    float*  Yf    = (float*)(S + 4 * szXh + szVt + szPr);
    __bf16* h1    = (__bf16*)(S);                 // [M,FF] bf16, overlaps Qh/Kh/Vh/Vt

    // per-layer Wt element offsets
    const size_t lstride = 4 * (size_t)kH * kH + 2 * (size_t)kH * kFF;   // 786432
    const size_t oQ = 0, oK = (size_t)kH * kH, oV = 2 * (size_t)kH * kH,
                 oO = 3 * (size_t)kH * kH, o1 = 4 * (size_t)kH * kH,
                 o2 = 4 * (size_t)kH * kH + (size_t)kH * kFF;

    auto tgrid = [](int elems) { return dim3((unsigned)((elems + 255) / 256)); };

    // ---- 1. pack weights (fp32 [K,N] -> bf16 [N,K]) ----
    for (int i = 0; i < kNL; ++i) {
        __bf16* base = WtAll + (size_t)i * lstride;
        transpose_w_kernel<<<tgrid(kH * kH),  256, 0, stream>>>(Wq + (size_t)i * kH * kH,  base + oQ, kH, kH);
        transpose_w_kernel<<<tgrid(kH * kH),  256, 0, stream>>>(Wk + (size_t)i * kH * kH,  base + oK, kH, kH);
        transpose_w_kernel<<<tgrid(kH * kH),  256, 0, stream>>>(Wv + (size_t)i * kH * kH,  base + oV, kH, kH);
        transpose_w_kernel<<<tgrid(kH * kH),  256, 0, stream>>>(Wo + (size_t)i * kH * kH,  base + oO, kH, kH);
        transpose_w_kernel<<<tgrid(kH * kFF), 256, 0, stream>>>(W1 + (size_t)i * kH * kFF, base + o1, kH, kFF);
        transpose_w_kernel<<<tgrid(kFF * kH), 256, 0, stream>>>(W2 + (size_t)i * kFF * kH, base + o2, kFF, kH);
    }

    // ---- 2. embedding + LN ----
    embed_ln_kernel<<<kM, 256, 0, stream>>>(input_ids, item_emb, pos_emb,
                                            emb_ln_w, emb_ln_b, X, Xh);

    // ---- 3. encoder layers ----
    const dim3 gH (kM / 32, kH  / (32 * GEMM_WAVES));   // (1600, 2)
    const dim3 gFF(kM / 32, kFF / (32 * GEMM_WAVES));   // (1600, 8)
    for (int i = 0; i < kNL; ++i) {
        const __bf16* base = WtAll + (size_t)i * lstride;
        // QKV projections (bf16 outputs)
        gemm_bias_kernel<<<gH, 32 * GEMM_WAVES, 0, stream>>>(
            Xh, base + oQ, bq + (size_t)i * kH, nullptr, Qh, kM, kH, kH, 0);
        gemm_bias_kernel<<<gH, 32 * GEMM_WAVES, 0, stream>>>(
            Xh, base + oK, bk + (size_t)i * kH, nullptr, Kh, kM, kH, kH, 0);
        gemm_bias_kernel<<<gH, 32 * GEMM_WAVES, 0, stream>>>(
            Xh, base + oV, bv + (size_t)i * kH, nullptr, Vh, kM, kH, kH, 0);
        // V repack for probs@V
        pack_vt_kernel<<<tgrid(kB * kNH * kDH * kLP), 256, 0, stream>>>(Vh, Vt);
        // scores + mask + softmax (WMMA QK^T)
        attn_scores_softmax_kernel<<<dim3(kLP / 16, kNH, kB), 32, 0, stream>>>(
            Qh, Kh, input_ids, probs);
        // ctx = probs @ V (WMMA)
        attn_ctx_kernel<<<dim3((kL + 15) / 16, kNH, kB), 32, 0, stream>>>(probs, Vt, ctx);
        // output projection (fp32 for residual) + LN
        gemm_bias_kernel<<<gH, 32 * GEMM_WAVES, 0, stream>>>(
            ctx, base + oO, bo + (size_t)i * kH, Yf, nullptr, kM, kH, kH, 0);
        add_ln_kernel<<<kM, 256, 0, stream>>>(Yf, X, attn_ln_w + (size_t)i * kH,
                                              attn_ln_b + (size_t)i * kH, X, Xh);
        // FFN
        gemm_bias_kernel<<<gFF, 32 * GEMM_WAVES, 0, stream>>>(
            Xh, base + o1, b1 + (size_t)i * kFF, nullptr, h1, kM, kFF, kH, 1); // GELU
        gemm_bias_kernel<<<gH, 32 * GEMM_WAVES, 0, stream>>>(
            h1, base + o2, b2 + (size_t)i * kH, Yf, nullptr, kM, kH, kFF, 0);
        add_ln_kernel<<<kM, 256, 0, stream>>>(Yf, X, ffn_ln_w + (size_t)i * kH,
                                              ffn_ln_b + (size_t)i * kH, X, Xh);
    }

    // ---- 4. output copy ----
    hipMemcpyAsync(out, X, (size_t)kM * kH * sizeof(float),
                   hipMemcpyDeviceToDevice, stream);
}
